// FeatureTransformer_43714177139318
// MI455X (gfx1250) — matrix operand order repaired and checked by reading, not compiled, not run on hardware
//
#include <hip/hip_runtime.h>
#include <hip/hip_bf16.h>
#include <cstdint>

// ---------------------------------------------------------------------------
// Shapes (fixed by the reference)
// ---------------------------------------------------------------------------
#define NB    8
#define LSEQ  4096
#define CDIM  256
#define C2DIM 512
#define HEADS 8
#define HDIM  32
#define NLROWS (NB * LSEQ)        // 32768

typedef __attribute__((ext_vector_type(16))) __bf16 bf16x16;
typedef __attribute__((ext_vector_type(8)))  float  f32x8;

union FragU { uint4 u[2]; bf16x16 v; };

// ---------------------------------------------------------------------------
// Generic bf16 WMMA GEMM:  C[M,N] = A[M,K] (row major, lda) x Bt[N,K]^T
// Bt is the weight stored TRANSPOSED (row n = output column, K contiguous),
// so both A and B wave fragments are contiguous 16B LDS reads.
// Block tile 128x128, BK=32, 8 waves -> each wave 32(M) x 64(N) = 2x4 WMMA tiles.
// EPI: 0 = f32 out, 1 = bf16 out, 2 = relu->bf16, 3 = elu(x)+1 -> bf16
// ---------------------------------------------------------------------------
template<int EPI>
__global__ __launch_bounds__(256)
void gemm_bf16_kernel(const __bf16* __restrict__ A, int lda, long long sA,
                      const __bf16* __restrict__ Bt, long long sB,
                      void* __restrict__ Cp, int ldc, long long sC,
                      int K)
{
    __shared__ __bf16 lA[128 * 32];
    __shared__ __bf16 lB[128 * 32];

    const __bf16* Ab = A  + (size_t)blockIdx.z * sA;
    const __bf16* Bb = Bt + (size_t)blockIdx.z * sB;

    const int t     = threadIdx.x;
    const int lane  = t & 31;
    const int wave  = t >> 5;
    const int wm    = wave >> 1;          // 0..3 (M direction)
    const int wn    = wave & 1;           // 0..1 (N direction)
    const int m0    = blockIdx.y * 128;
    const int n0    = blockIdx.x * 128;
    const int lrow  = t >> 1;             // 0..127 (tile loader row)
    const int lhalf = t & 1;              // which 32B half of the 64B row

    f32x8 acc[2][4] = {};

    const uint4* gA = (const uint4*)(Ab + (size_t)(m0 + lrow) * lda + lhalf * 16);
    const uint4* gB = (const uint4*)(Bb + (size_t)(n0 + lrow) * K   + lhalf * 16);
    uint4* sA4 = (uint4*)lA + lrow * 4 + lhalf * 2;
    uint4* sB4 = (uint4*)lB + lrow * 4 + lhalf * 2;

    const int ar  = lane & 15;
    const int ac0 = (lane < 16) ? 0 : 8;      // A: K chunks {c0..c0+7, c0+16..c0+23}
    const int bk0 = (lane < 16) ? 0 : 8;      // B: K chunks {c0..c0+7, c0+16..c0+23}

    for (int kt = 0; kt < K; kt += 32) {
        uint4 a0 = gA[0], a1 = gA[1];
        uint4 b0 = gB[0], b1 = gB[1];
        gA += 4; gB += 4;                     // advance 32 bf16 along the row
        if (kt + 32 < K) {                    // gfx1250 global_prefetch_b8
            __builtin_prefetch((const void*)gA, 0, 3);
            __builtin_prefetch((const void*)gB, 0, 3);
        }
        sA4[0] = a0; sA4[1] = a1;
        sB4[0] = b0; sB4[1] = b1;
        __syncthreads();

        FragU af[2], bfr[4];
        #pragma unroll
        for (int i = 0; i < 2; ++i) {
            const __bf16* p = lA + (wm * 32 + i * 16 + ar) * 32;
            af[i].u[0] = *(const uint4*)(p + ac0);
            af[i].u[1] = *(const uint4*)(p + ac0 + 16);
        }
        #pragma unroll
        for (int j = 0; j < 4; ++j) {
            const __bf16* p = lB + (wn * 64 + j * 16 + ar) * 32 + bk0;
            bfr[j].u[0] = *(const uint4*)(p);
            bfr[j].u[1] = *(const uint4*)(p + 16);
        }
        #pragma unroll
        for (int i = 0; i < 2; ++i)
            #pragma unroll
            for (int j = 0; j < 4; ++j)
                acc[i][j] = __builtin_amdgcn_wmma_f32_16x16x32_bf16(
                    false, af[i].v, false, bfr[j].v, (short)0, acc[i][j],
                    false, false);
        __syncthreads();
    }

    // Epilogue: VGPR r holds row (r or r+8 by lane half), col = lane%16
    float*  Cf = (float*)Cp  + (size_t)blockIdx.z * sC;
    __bf16* Cb = (__bf16*)Cp + (size_t)blockIdx.z * sC;
    #pragma unroll
    for (int i = 0; i < 2; ++i) {
        #pragma unroll
        for (int j = 0; j < 4; ++j) {
            const int rbase = m0 + wm * 32 + i * 16 + ((lane < 16) ? 0 : 8);
            const int col   = n0 + wn * 64 + j * 16 + (lane & 15);
            #pragma unroll
            for (int r = 0; r < 8; ++r) {
                float v = acc[i][j][r];
                size_t idx = (size_t)(rbase + r) * ldc + col;
                if (EPI == 0) {
                    Cf[idx] = v;
                } else {
                    if (EPI == 2) v = v > 0.f ? v : 0.f;
                    if (EPI == 3) v = v > 0.f ? v + 1.f : __expf(v);
                    Cb[idx] = (__bf16)v;
                }
            }
        }
    }
}

// ---------------------------------------------------------------------------
// Weight transpose + bf16 convert: Wt[n*K + k] = W[k*N + n]
// ---------------------------------------------------------------------------
__global__ void w_transpose_bf16(const float* __restrict__ W,
                                 __bf16* __restrict__ Wt, int K, int N)
{
    int gid = blockIdx.x * 256 + threadIdx.x;
    if (gid >= K * N) return;
    int k = gid / N, n = gid - k * N;
    Wt[(size_t)n * K + k] = (__bf16)W[gid];
}

// f32 [NL,256] -> bf16 into cat first half (row stride 512)
__global__ void pack_cat_kernel(const float* __restrict__ x, __bf16* __restrict__ cat)
{
    int gid = blockIdx.x * 256 + threadIdx.x;
    int i = gid * 4, row = i >> 8, c = i & 255;
    float4 v = *(const float4*)(x + i);
    __bf16* o = cat + (size_t)row * 512 + c;
    o[0] = (__bf16)v.x; o[1] = (__bf16)v.y; o[2] = (__bf16)v.z; o[3] = (__bf16)v.w;
}

// f32 -> bf16 contiguous
__global__ void pack_kernel(const float* __restrict__ x, __bf16* __restrict__ o)
{
    int gid = blockIdx.x * 256 + threadIdx.x;
    int i = gid * 4;
    float4 v = *(const float4*)(x + i);
    __bf16* p = o + i;
    p[0] = (__bf16)v.x; p[1] = (__bf16)v.y; p[2] = (__bf16)v.z; p[3] = (__bf16)v.w;
}

__global__ void fill0_bf16(__bf16* p)
{
    int gid = blockIdx.x * 256 + threadIdx.x;
    ((uint2*)p)[gid] = make_uint2(0u, 0u);     // 4 bf16 per thread
}

// ---------------------------------------------------------------------------
// Per (n,h): KVt[e][d] = sum_s K[s,d]*V[s,e]; Ksum[d] = sum_s K[s,d]
// KVt written into the block-diagonal B matrix (bkv, pre-zeroed).
// ---------------------------------------------------------------------------
__global__ __launch_bounds__(256)
void kv_kernel(const __bf16* __restrict__ kf, const __bf16* __restrict__ vf,
               __bf16* __restrict__ bkv, float* __restrict__ ksum)
{
    __shared__ __bf16 sk[64 * 32];
    __shared__ __bf16 sv[64 * 32];
    const int n = blockIdx.x >> 3, h = blockIdx.x & 7;
    const int t = threadIdx.x;
    const int e = t >> 3, dbase = (t & 7) * 4;
    float acc[4] = {0, 0, 0, 0}, kss[4] = {0, 0, 0, 0};

    const int lr = (t & 127) >> 1, lhalf = t & 1;
    const __bf16* gsrc = (t < 128) ? kf : vf;
    __bf16*       ldst = (t < 128) ? sk : sv;

    for (int s0 = 0; s0 < LSEQ; s0 += 64) {
        const uint4* g = (const uint4*)(gsrc +
            ((size_t)(n * LSEQ + s0 + lr)) * 256 + h * 32 + lhalf * 16);
        uint4 d0 = g[0], d1 = g[1];
        uint4* sd = (uint4*)(ldst + lr * 32 + lhalf * 16);
        sd[0] = d0; sd[1] = d1;
        __syncthreads();
        #pragma unroll 4
        for (int s = 0; s < 64; ++s) {
            float vv = (float)sv[s * 32 + e];
            #pragma unroll
            for (int c = 0; c < 4; ++c) {
                float kk = (float)sk[s * 32 + dbase + c];
                acc[c] += kk * vv;
                if (e == 0) kss[c] += kk;
            }
        }
        __syncthreads();
    }
    __bf16* out = bkv + (size_t)n * 65536 + (size_t)(h * 32 + e) * 256 + h * 32 + dbase;
    #pragma unroll
    for (int c = 0; c < 4; ++c) out[c] = (__bf16)acc[c];
    if (e == 0) {
        float* kso = ksum + (n * 8 + h) * 32 + dbase;
        #pragma unroll
        for (int c = 0; c < 4; ++c) kso[c] = kss[c];
    }
}

// Fold Z = 1/(Q.Ksum + eps) into Q rows (in place). One thread per (row, head).
__global__ void zq_kernel(__bf16* __restrict__ qf, const float* __restrict__ ksum)
{
    int gid = blockIdx.x * 256 + threadIdx.x;
    int row = gid >> 3, h = gid & 7;
    int n = row >> 12;                         // L = 4096
    __bf16* q = qf + (size_t)row * 256 + h * 32;
    const float* ks = ksum + (n * 8 + h) * 32;
    float qv[32], dot = 0.f;
    #pragma unroll
    for (int c = 0; c < 32; ++c) { qv[c] = (float)q[c]; dot += qv[c] * ks[c]; }
    float z = 1.f / (dot + 1e-6f);
    #pragma unroll
    for (int c = 0; c < 32; ++c) q[c] = (__bf16)(qv[c] * z);
}

// ---------------------------------------------------------------------------
// LayerNorm (one wave32 per row of 256). Variant 1: -> bf16 into cat[:,256:].
// Variant 2: feat = feat + LN(msg) (f32, in place residual).
// ---------------------------------------------------------------------------
__device__ __forceinline__ void ln_row(const float* x, float xs[8],
                                       float& mean, float& rstd, int lane)
{
    const float4 v0 = ((const float4*)(x + lane * 8))[0];
    const float4 v1 = ((const float4*)(x + lane * 8))[1];
    xs[0] = v0.x; xs[1] = v0.y; xs[2] = v0.z; xs[3] = v0.w;
    xs[4] = v1.x; xs[5] = v1.y; xs[6] = v1.z; xs[7] = v1.w;
    float s = 0.f, q = 0.f;
    #pragma unroll
    for (int u = 0; u < 8; ++u) { s += xs[u]; q += xs[u] * xs[u]; }
    #pragma unroll
    for (int o = 16; o > 0; o >>= 1) {
        s += __shfl_xor(s, o);
        q += __shfl_xor(q, o);
    }
    mean = s * (1.f / 256.f);
    float var = q * (1.f / 256.f) - mean * mean;
    rstd = rsqrtf(var + 1e-5f);
}

__global__ __launch_bounds__(256)
void ln_to_cat_kernel(const float* __restrict__ msg, const float* __restrict__ g,
                      const float* __restrict__ b, __bf16* __restrict__ cat)
{
    const int row  = blockIdx.x * 8 + (threadIdx.x >> 5);
    const int lane = threadIdx.x & 31;
    float xs[8], mean, rstd;
    ln_row(msg + (size_t)row * 256, xs, mean, rstd, lane);
    __bf16* o = cat + (size_t)row * 512 + 256 + lane * 8;
    #pragma unroll
    for (int u = 0; u < 8; ++u) {
        int c = lane * 8 + u;
        o[u] = (__bf16)((xs[u] - mean) * rstd * g[c] + b[c]);
    }
}

__global__ __launch_bounds__(256)
void ln_res_kernel(const float* __restrict__ msg, const float* __restrict__ g,
                   const float* __restrict__ b, float* __restrict__ feat)
{
    const int row  = blockIdx.x * 8 + (threadIdx.x >> 5);
    const int lane = threadIdx.x & 31;
    float xs[8], mean, rstd;
    ln_row(msg + (size_t)row * 256, xs, mean, rstd, lane);
    float* o = feat + (size_t)row * 256 + lane * 8;
    #pragma unroll
    for (int u = 0; u < 8; ++u) {
        int c = lane * 8 + u;
        o[u] = o[u] + (xs[u] - mean) * rstd * g[c] + b[c];
    }
}

// ---------------------------------------------------------------------------
// Host orchestration
// ---------------------------------------------------------------------------
extern "C" void kernel_launch(void* const* d_in, const int* in_sizes, int n_in,
                              void* d_out, int out_size, void* d_ws, size_t ws_size,
                              hipStream_t stream)
{
    (void)in_sizes; (void)n_in; (void)out_size; (void)ws_size;
    const int C = CDIM, C2 = C2DIM, NL = NLROWS;
    const size_t NLC = (size_t)NL * C;

    float* feat0 = (float*)d_out;
    float* feat1 = feat0 + NLC;
    hipMemcpyAsync(feat0, d_in[0], NLC * sizeof(float), hipMemcpyDeviceToDevice, stream);
    hipMemcpyAsync(feat1, d_in[1], NLC * sizeof(float), hipMemcpyDeviceToDevice, stream);

    // ---- workspace carve ----
    char* p = (char*)d_ws;
    auto carve = [&](size_t bytes) -> char* {
        char* r = p; p += (bytes + 255) & ~(size_t)255; return r;
    };
    const size_t LWT = 655360;   // bf16 elems per layer of transposed weights
    __bf16* WT   = (__bf16*)carve((size_t)8 * LWT * 2);
    __bf16* cat  = (__bf16*)carve((size_t)NL * 512 * 2);   // [x | LN(msg)] bf16
    __bf16* sb   = (__bf16*)carve(NLC * 2);                // source bf16
    __bf16* qf   = (__bf16*)carve(NLC * 2);                // Q feature (then Qz)
    __bf16* kf   = (__bf16*)carve(NLC * 2);                // K feature / attn out
    __bf16* vf   = (__bf16*)carve(NLC * 2);
    float*  msg  = (float*)carve(NLC * 4);
    __bf16* hb   = (__bf16*)carve((size_t)NL * 512 * 2);
    __bf16* bkv  = (__bf16*)carve((size_t)8 * 65536 * 2);  // block-diag KV^T
    float*  ksum = (float*)carve((size_t)64 * 32 * 4);
    __bf16* at   = kf;  // K-feature is dead after kv_kernel; reuse for attn out

    const float* Wq = (const float*)d_in[2];
    const float* Wk = (const float*)d_in[3];
    const float* Wv = (const float*)d_in[4];
    const float* Wm = (const float*)d_in[5];
    const float* W1 = (const float*)d_in[6];
    const float* W2 = (const float*)d_in[7];
    const float* g1 = (const float*)d_in[8];
    const float* b1 = (const float*)d_in[9];
    const float* g2 = (const float*)d_in[10];
    const float* b2 = (const float*)d_in[11];

    // ---- convert + transpose all layer weights to bf16 once ----
    auto tr = [&](const float* W, __bf16* dst, int K, int N) {
        int tot = K * N;
        w_transpose_bf16<<<(tot + 255) / 256, 256, 0, stream>>>(W, dst, K, N);
    };
    for (int l = 0; l < 8; ++l) {
        __bf16* base = WT + (size_t)l * LWT;
        tr(Wq + (size_t)l * C * C,   base + 0,      C,  C);
        tr(Wk + (size_t)l * C * C,   base + 65536,  C,  C);
        tr(Wv + (size_t)l * C * C,   base + 131072, C,  C);
        tr(Wm + (size_t)l * C * C,   base + 196608, C,  C);
        tr(W1 + (size_t)l * C2 * C2, base + 262144, C2, C2);
        tr(W2 + (size_t)l * C2 * C,  base + 524288, C2, C);
    }

    auto gemm = [&](int epi, const __bf16* A, int lda, long long sA,
                    const __bf16* B, long long sB,
                    void* Cp, int ldc, long long sC,
                    int Mper, int Nn, int Kk, int batches) {
        dim3 grid(Nn / 128, Mper / 128, batches), blk(256);
        switch (epi) {
        case 0: gemm_bf16_kernel<0><<<grid, blk, 0, stream>>>(A, lda, sA, B, sB, Cp, ldc, sC, Kk); break;
        case 1: gemm_bf16_kernel<1><<<grid, blk, 0, stream>>>(A, lda, sA, B, sB, Cp, ldc, sC, Kk); break;
        case 2: gemm_bf16_kernel<2><<<grid, blk, 0, stream>>>(A, lda, sA, B, sB, Cp, ldc, sC, Kk); break;
        default: gemm_bf16_kernel<3><<<grid, blk, 0, stream>>>(A, lda, sA, B, sB, Cp, ldc, sC, Kk); break;
        }
    };

    auto update = [&](float* x, const float* src, int l) {
        __bf16* wqT = WT + (size_t)l * LWT;
        __bf16* wkT = wqT + 65536;
        __bf16* wvT = wqT + 131072;
        __bf16* wmT = wqT + 196608;
        __bf16* w1T = wqT + 262144;
        __bf16* w2T = wqT + 524288;
        const float* g1l = g1 + (size_t)l * C;
        const float* b1l = b1 + (size_t)l * C;
        const float* g2l = g2 + (size_t)l * C;
        const float* b2l = b2 + (size_t)l * C;

        pack_cat_kernel<<<(int)(NLC / 4 / 256), 256, 0, stream>>>(x, cat);
        pack_kernel<<<(int)(NLC / 4 / 256), 256, 0, stream>>>(src, sb);

        gemm(3, cat, 512, 0, wqT, 0, qf, 256, 0, NL, 256, 256, 1);   // Q = elu(xWq)+1
        gemm(3, sb,  256, 0, wkT, 0, kf, 256, 0, NL, 256, 256, 1);   // K = elu(sWk)+1
        gemm(1, sb,  256, 0, wvT, 0, vf, 256, 0, NL, 256, 256, 1);   // V = sWv

        fill0_bf16<<<(8 * 65536 / 4) / 256, 256, 0, stream>>>(bkv);
        kv_kernel<<<64, 256, 0, stream>>>(kf, vf, bkv, ksum);
        zq_kernel<<<(NL * 8) / 256, 256, 0, stream>>>(qf, ksum);

        // attn = Qz @ blockdiag(KV), batched over n
        gemm(1, qf, 256, (long long)LSEQ * C, bkv, 65536,
             at, 256, (long long)LSEQ * C, LSEQ, 256, 256, 8);

        gemm(0, at, 256, 0, wmT, 0, msg, 256, 0, NL, 256, 256, 1);   // message = attn@Wm (f32)
        ln_to_cat_kernel<<<NL / 8, 256, 0, stream>>>(msg, g1l, b1l, cat);

        gemm(2, cat, 512, 0, w1T, 0, hb, 512, 0, NL, 512, 512, 1);   // relu(cat@W1)
        gemm(0, hb,  512, 0, w2T, 0, msg, 256, 0, NL, 256, 512, 1);  // @W2 (f32)
        ln_res_kernel<<<NL / 8, 256, 0, stream>>>(msg, g2l, b2l, x); // x += LN(m2)
    };

    // LAYER_NAMES = ['self','cross'] * 4
    for (int i = 0; i < 8; ++i) {
        if ((i & 1) == 0) {               // self
            update(feat0, feat0, i);
            update(feat1, feat1, i);
        } else {                          // cross (feat1 sees updated feat0)
            update(feat0, feat1, i);
            update(feat1, feat0, i);
        }
    }
}
